// Attention_5566277616422
// MI455X (gfx1250) — compile-verified, matrix operand order checked by reading, not run on hardware
//
#include <hip/hip_runtime.h>
#include <hip/hip_bf16.h>
#include <math.h>

// ---------------------------------------------------------------------------
// Types for CDNA5 WMMA (wave32): A/B = 16 x bf16, C/D = 8 x f32 per lane.
// ---------------------------------------------------------------------------
typedef __attribute__((ext_vector_type(16))) __bf16 v16bf;
typedef __attribute__((ext_vector_type(8)))  __bf16 v8bf;
typedef __attribute__((ext_vector_type(8)))  float  v8f;

#define BB   4
#define SS   1024
#define DIMM 1024
#define HH   16
#define DHH  64
#define RR   (BB * SS)          // 4096 token rows

static __device__ __forceinline__ __bf16 f2bf(float f) {
    unsigned u = __builtin_bit_cast(unsigned, f);
    unsigned r = u + 0x7fffu + ((u >> 16) & 1u);   // round-to-nearest-even
    unsigned short h = (unsigned short)(r >> 16);
    return __builtin_bit_cast(__bf16, h);
}

// ---------------------------------------------------------------------------
// Pure-VALU 16-lane reductions via DPP (no LDS, no dscnt waits).
// Masks {xor1, xor2, xor7(row_half_mirror), xor15(row_mirror)} generate the
// full 16-lane group; wave32 DPP rows are 16 lanes, so the two lane-halves
// (which hold different C-layout rows) reduce independently — exactly right.
// ---------------------------------------------------------------------------
template <int CTRL>
static __device__ __forceinline__ float dpp_mov(float x) {
    int s = __builtin_bit_cast(int, x);
    int r = __builtin_amdgcn_update_dpp(s, s, CTRL, 0xF, 0xF, false);
    return __builtin_bit_cast(float, r);
}
static __device__ __forceinline__ float red16_max(float x) {
    x = fmaxf(x, dpp_mov<0xB1>(x));    // quad_perm [1,0,3,2]  (xor 1)
    x = fmaxf(x, dpp_mov<0x4E>(x));    // quad_perm [2,3,0,1]  (xor 2)
    x = fmaxf(x, dpp_mov<0x141>(x));   // row_half_mirror      (xor 7)
    x = fmaxf(x, dpp_mov<0x140>(x));   // row_mirror           (xor 15)
    return x;
}
static __device__ __forceinline__ float red16_sum(float x) {
    x += dpp_mov<0xB1>(x);
    x += dpp_mov<0x4E>(x);
    x += dpp_mov<0x141>(x);
    x += dpp_mov<0x140>(x);
    return x;
}

// ---------------------------------------------------------------------------
// Load a 16x32 bf16 fragment (A-layout; B-layout is the mirror with N per
// lane) from row-major memory with row stride `ld` (elements).
// Lane l -> row (l&15); K halves: lanes<16 get K 0-7 / 16-23,
// lanes>=16 get K 8-15 / 24-31.  All chunks 16B aligned if ld%8==0.
// ---------------------------------------------------------------------------
static __device__ __forceinline__ v16bf load_frag16x32(const __bf16* p0, int lane, int ld) {
    const __bf16* p = p0 + (size_t)(lane & 15) * ld + ((lane >> 4) & 1) * 8;
    v8bf lo = *(const v8bf*)(p);
    v8bf hi = *(const v8bf*)(p + 16);
    v16bf r;
#pragma unroll
    for (int i = 0; i < 8; ++i) { r[i] = lo[i]; r[i + 8] = hi[i]; }
    return r;
}

// ---------------------------------------------------------------------------
// fp32 -> bf16 convert
// ---------------------------------------------------------------------------
__global__ void cvt_f32_bf16(const float* __restrict__ src, __bf16* __restrict__ dst, int n) {
    int i = blockIdx.x * blockDim.x + threadIdx.x;
    if (i < n) dst[i] = f2bf(src[i]);
}

// ---------------------------------------------------------------------------
// GEMM: out[r][c] = sum_d A[r][d] * W[c][d] + bias[c]   (y = x @ W^T + b)
// Block = 256 threads = 8 waves, block tile M=128 x N=64.  All 8 waves share
// the same 64 output columns, so the W tile (64x32 bf16, rows padded to 40 to
// spread LDS banks) is staged in LDS once per k-step (8x L2 traffic cut).
// MODE 0: fp32 out (RR, DIMM)                           (final projection)
// MODE 1: RoPE + permute -> bf16 [B,H,S,DH]             (Q, K)
// MODE 2: permute -> bf16 [B,H,S,DH]                    (V)
// ---------------------------------------------------------------------------
#define BPAD 40

template <int MODE>
__global__ void gemm_wmma(const __bf16* __restrict__ A, const __bf16* __restrict__ W,
                          const float* __restrict__ bias,
                          float* __restrict__ outF, __bf16* __restrict__ outB) {
    __shared__ __align__(16) __bf16 ldsB[64 * BPAD];

    const int tid  = threadIdx.x;
    const int lane = tid & 31;
    const int wave = tid >> 5;
    const int n0 = blockIdx.x * 64;
    const int m0 = blockIdx.y * 128 + wave * 16;

    const int fn = tid >> 2;            // 0..63  column to stage
    const int fk = (tid & 3) * 8;       // 0,8,16,24

    v8f acc[4] = {};
    for (int k0 = 0; k0 < DIMM; k0 += 32) {
        __syncthreads();                // previous iteration's reads done
        *(v8bf*)&ldsB[fn * BPAD + fk] =
            *(const v8bf*)&W[(size_t)(n0 + fn) * DIMM + k0 + fk];
        __syncthreads();

        v16bf a = load_frag16x32(A + (size_t)m0 * DIMM + k0, lane, DIMM);
#pragma unroll
        for (int t = 0; t < 4; ++t) {
            v16bf b = load_frag16x32(ldsB + (t * 16) * BPAD, lane, BPAD);
            acc[t] = __builtin_amdgcn_wmma_f32_16x16x32_bf16(
                false, a, false, b, (short)0, acc[t], false, false);
        }
    }

    const int cn     = lane & 15;
    const int rowoff = ((lane >> 4) & 1) ? 8 : 0;

    if (MODE == 0) {
#pragma unroll
        for (int t = 0; t < 4; ++t) {
            int col = n0 + t * 16 + cn;
            float bv = bias[col];
#pragma unroll
            for (int i = 0; i < 8; ++i) {
                int row = m0 + i + rowoff;
                outF[(size_t)row * DIMM + col] = acc[t][i] + bv;
            }
        }
    } else if (MODE == 1) {
        // This wave's 64 columns are exactly one head; RoPE pair (d, d+32)
        // sits in acc[t] / acc[t+2] of the same lane.
        const int h = n0 / DHH;
#pragma unroll
        for (int t = 0; t < 2; ++t) {
            int d = t * 16 + cn;                       // [0, 32)
            float b1 = bias[n0 + d];
            float b2 = bias[n0 + d + 32];
            // freq = 10000^(-d/32) = exp(-d * ln(10000)/32)
            float freq = __expf(-(float)d * 0.28782313662425572f);
#pragma unroll
            for (int i = 0; i < 8; ++i) {
                int row = m0 + i + rowoff;             // row = b*S + s
                int bi = row >> 10;                    // / SS
                int s  = row & (SS - 1);
                float ang = (float)s * freq;
                float c = __cosf(ang), sn = __sinf(ang);
                float x1 = acc[t][i]     + b1;
                float x2 = acc[t + 2][i] + b2;
                size_t o = ((size_t)(bi * HH + h) * SS + s) * DHH;
                outB[o + d]      = f2bf(x1 * c - x2 * sn);
                outB[o + d + 32] = f2bf(x2 * c + x1 * sn);
            }
        }
    } else {
        const int h = n0 / DHH;
#pragma unroll
        for (int t = 0; t < 4; ++t) {
            int d = t * 16 + cn;
            float bv = bias[n0 + d];
#pragma unroll
            for (int i = 0; i < 8; ++i) {
                int row = m0 + i + rowoff;
                int bi = row >> 10;
                int s  = row & (SS - 1);
                outB[((size_t)(bi * HH + h) * SS + s) * DHH + d] = f2bf(acc[t][i] + bv);
            }
        }
    }
}

// ---------------------------------------------------------------------------
// Flash attention (online softmax), wave-granular, block-cooperative staging.
// All 8 waves of a block share one (b,h)  (8 | 64 q-tiles per head), so K and
// V tiles (32 kv rows) are staged in LDS once per step for the whole block.
// V is stored transposed [dh][kv] so the P@V B-fragment is two ds_read_b128.
// Softmax row stats use pure-DPP 16-lane reductions (no LDS round trips).
// Row pads (72 / 40) break bank conflicts; every 16B chunk stays aligned.
// ---------------------------------------------------------------------------
#define KPAD 72
#define VPAD 40
#define PPAD 40

__global__ void flash_attn(const __bf16* __restrict__ Q, const __bf16* __restrict__ K,
                           const __bf16* __restrict__ V, __bf16* __restrict__ O) {
    __shared__ __align__(16) __bf16 ldsK[32 * KPAD];      // [kv][dh]
    __shared__ __align__(16) __bf16 ldsVT[64 * VPAD];     // [dh][kv]
    __shared__ __align__(16) __bf16 ldsP[8][16 * PPAD];   // per-wave P tile

    const int tid  = threadIdx.x;
    const int lane = tid & 31;
    const int wave = tid >> 5;
    const int bh = blockIdx.x >> 3;                 // block shares one (b,h)
    const int qt = (blockIdx.x & 7) * 8 + wave;     // q-tile within head
    const int b = bh >> 4, h = bh & 15;
    const int m0 = qt * 16;

    const __bf16* Qp = Q + (size_t)bh * SS * DHH;
    const __bf16* Kp = K + (size_t)bh * SS * DHH;
    const __bf16* Vp = V + (size_t)bh * SS * DHH;

    v16bf aq0 = load_frag16x32(Qp + (size_t)m0 * DHH + 0,  lane, DHH);
    v16bf aq1 = load_frag16x32(Qp + (size_t)m0 * DHH + 32, lane, DHH);

    v8f o[4] = {};
    float mi[8], li[8];
#pragma unroll
    for (int i = 0; i < 8; ++i) { mi[i] = -1e30f; li[i] = 0.0f; }

    const int cn     = lane & 15;
    const int rowoff = ((lane >> 4) & 1) ? 8 : 0;
    const float scale = 0.125f;                     // 1/sqrt(DH)

    const int fkv = tid >> 3;                       // 0..31   staging row
    const int fc0 = (tid & 7) * 8;                  // 0..56   staging chunk

    for (int j = 0; j < SS; j += 32) {
        // ---- cooperative stage of K (row-major) and V (transposed) ------
        __syncthreads();                            // previous reads done
        *(v8bf*)&ldsK[fkv * KPAD + fc0] =
            *(const v8bf*)&Kp[(size_t)(j + fkv) * DHH + fc0];
        v8bf vv = *(const v8bf*)&Vp[(size_t)(j + fkv) * DHH + fc0];
#pragma unroll
        for (int e = 0; e < 8; ++e) ldsVT[(fc0 + e) * VPAD + fkv] = vv[e];
        __syncthreads();

        // ---- S tile (16 x 32) = Q(16x64) . K^T(64x32) --------------------
        v8f sfr[2];
#pragma unroll
        for (int t = 0; t < 2; ++t) {
            v16bf bk0 = load_frag16x32(ldsK + (t * 16) * KPAD + 0,  lane, KPAD);
            v16bf bk1 = load_frag16x32(ldsK + (t * 16) * KPAD + 32, lane, KPAD);
            v8f z = {};
            z = __builtin_amdgcn_wmma_f32_16x16x32_bf16(false, aq0, false, bk0, (short)0, z, false, false);
            z = __builtin_amdgcn_wmma_f32_16x16x32_bf16(false, aq1, false, bk1, (short)0, z, false, false);
            sfr[t] = z;
        }
#pragma unroll
        for (int t = 0; t < 2; ++t)
#pragma unroll
            for (int i = 0; i < 8; ++i) sfr[t][i] *= scale;

        // ---- online softmax stats (pure-DPP reductions) ------------------
        float mnew[8], alpha[8];
#pragma unroll
        for (int i = 0; i < 8; ++i) {
            float mx = red16_max(fmaxf(sfr[0][i], sfr[1][i]));
            float mn = fmaxf(mi[i], mx);
            mnew[i]  = mn;
            alpha[i] = __expf(mi[i] - mn);
        }
#pragma unroll
        for (int i = 0; i < 8; ++i) {
            float p0 = __expf(sfr[0][i] - mnew[i]);
            float p1 = __expf(sfr[1][i] - mnew[i]);
            sfr[0][i] = p0; sfr[1][i] = p1;
            li[i] = li[i] * alpha[i] + red16_sum(p0 + p1);
            mi[i] = mnew[i];
        }
#pragma unroll
        for (int t = 0; t < 4; ++t)
#pragma unroll
            for (int i = 0; i < 8; ++i) o[t][i] *= alpha[i];

        // ---- P: C-layout -> LDS -> A-layout ------------------------------
#pragma unroll
        for (int t = 0; t < 2; ++t)
#pragma unroll
            for (int i = 0; i < 8; ++i)
                ldsP[wave][(i + rowoff) * PPAD + t * 16 + cn] = f2bf(sfr[t][i]);
        __syncthreads();
        v16bf pf = load_frag16x32(&ldsP[wave][0], lane, PPAD);

        // ---- O += P(16x32) @ V(32x64)  (B frags from transposed LDS V) ---
#pragma unroll
        for (int t = 0; t < 4; ++t) {
            v16bf bv = load_frag16x32(ldsVT + (t * 16) * VPAD, lane, VPAD);
            o[t] = __builtin_amdgcn_wmma_f32_16x16x32_bf16(false, pf, false, bv, (short)0, o[t], false, false);
        }
    }

    // ---- normalize and store bf16 into [B, S, DIM] ----------------------
#pragma unroll
    for (int i = 0; i < 8; ++i) {
        float inv = 1.0f / li[i];
        int s = m0 + i + rowoff;
#pragma unroll
        for (int t = 0; t < 4; ++t) {
            int col = h * DHH + t * 16 + cn;
            O[(size_t)(b * SS + s) * DIMM + col] = f2bf(o[t][i] * inv);
        }
    }
}

// ---------------------------------------------------------------------------
// Launch
// ---------------------------------------------------------------------------
extern "C" void kernel_launch(void* const* d_in, const int* in_sizes, int n_in,
                              void* d_out, int out_size, void* d_ws, size_t ws_size,
                              hipStream_t stream) {
    const float* x  = (const float*)d_in[0];
    const float* Wq = (const float*)d_in[1];
    const float* bq = (const float*)d_in[2];
    const float* Wk = (const float*)d_in[3];
    const float* bk = (const float*)d_in[4];
    const float* Wv = (const float*)d_in[5];
    const float* bv = (const float*)d_in[6];
    const float* Wo = (const float*)d_in[7];
    const float* bo = (const float*)d_in[8];
    float* out = (float*)d_out;

    char* ws = (char*)d_ws;
    __bf16* xb  = (__bf16*)(ws);                       // 8 MB  x bf16
    __bf16* wqb = (__bf16*)(ws + (8u  << 20));         // 2 MB each weight
    __bf16* wkb = (__bf16*)(ws + (10u << 20));
    __bf16* wvb = (__bf16*)(ws + (12u << 20));
    __bf16* wob = (__bf16*)(ws + (14u << 20));
    __bf16* qb  = (__bf16*)(ws + (16u << 20));         // 8 MB  [B,H,S,DH]
    __bf16* kb  = (__bf16*)(ws + (24u << 20));
    __bf16* vb2 = (__bf16*)(ws + (32u << 20));
    __bf16* ob  = (__bf16*)(ws + (40u << 20));         // 8 MB  [B,S,DIM]
    // total workspace use: 48 MB

    const int nX = RR * DIMM;
    const int nW = DIMM * DIMM;
    cvt_f32_bf16<<<(nX + 255) / 256, 256, 0, stream>>>(x,  xb,  nX);
    cvt_f32_bf16<<<(nW + 255) / 256, 256, 0, stream>>>(Wq, wqb, nW);
    cvt_f32_bf16<<<(nW + 255) / 256, 256, 0, stream>>>(Wk, wkb, nW);
    cvt_f32_bf16<<<(nW + 255) / 256, 256, 0, stream>>>(Wv, wvb, nW);
    cvt_f32_bf16<<<(nW + 255) / 256, 256, 0, stream>>>(Wo, wob, nW);

    dim3 g(DIMM / 64, RR / 128);                       // exact tiling, EXEC all-1s
    gemm_wmma<1><<<g, 256, 0, stream>>>(xb, wqb, bq, nullptr, qb);
    gemm_wmma<1><<<g, 256, 0, stream>>>(xb, wkb, bk, nullptr, kb);
    gemm_wmma<2><<<g, 256, 0, stream>>>(xb, wvb, bv, nullptr, vb2);

    flash_attn<<<(BB * HH * (SS / 16)) / 8, 256, 0, stream>>>(qb, kb, vb2, ob);

    gemm_wmma<0><<<g, 256, 0, stream>>>(ob, wob, bo, out, nullptr);
}
